// PointNetSetAbstractionEdgeSA_20366734918002
// MI455X (gfx1250) — matrix operand order repaired and checked by reading, not compiled, not run on hardware
//
#include <hip/hip_runtime.h>
#include <hip/hip_bf16.h>

typedef __attribute__((ext_vector_type(16))) _Float16 v16h;
typedef __attribute__((ext_vector_type(8)))  _Float16 v8h;
typedef __attribute__((ext_vector_type(8)))  float    v8f;

#define B_   8
#define N_   8192
#define S_   2048
#define K_   32
#define D_   64
#define DM_  256
#define NH_  4
#define HD_  64
#define M_   (B_*S_*K_)      // 524288
#define BS_  (B_*S_)         // 16384
#define CP1_ 160             // 131 padded to mult of 32

// ---- weight repack fp32 -> fp16, directly into WMMA B-fragment order -------
// Wp layout: block = (kblk * (cols/16) + tn), 512 halves per block:
//   Wp[block*512 + lane*16 + e] = W[kblk*32 + (lane>>4)*16 + e, tn*16 + (lane&15)]
__global__ void cvt_w_pack(const float* __restrict__ W, _Float16* __restrict__ Wp,
                           int rows, int rowsPad, int cols) {
  int i = blockIdx.x * blockDim.x + threadIdx.x;
  int total = rowsPad * cols;
  if (i >= total) return;
  int e    = i & 15;
  int lane = (i >> 4) & 31;
  int blk  = i >> 9;
  int tilesN = cols >> 4;
  int kblk = blk / tilesN, tn = blk % tilesN;
  int h = lane >> 4, n = lane & 15;
  int k = kblk * 32 + h * 16 + e;
  int c = tn * 16 + n;
  Wp[i] = (k < rows) ? (_Float16)W[(size_t)k * cols + c] : (_Float16)0.f;
}

// ---------------- farthest point sampling: 1 block per batch ----------------
__global__ void fps_kernel(const float* __restrict__ xyz, int* __restrict__ fps_idx) {
  int b = blockIdx.x;
  __shared__ float dist[N_];
  __shared__ float rv[1024];
  __shared__ int   ri[1024];
  __shared__ int   s_far;
  const float* xb = xyz + (size_t)b * 3 * N_;
  int t = threadIdx.x;
  for (int n = t; n < N_; n += blockDim.x) dist[n] = 1e10f;
  if (t == 0) s_far = 0;
  __syncthreads();
  for (int s = 0; s < S_; ++s) {
    int far = s_far;
    if (t == 0) fps_idx[b * S_ + s] = far;
    float cx = xb[far], cy = xb[N_ + far], cz = xb[2 * N_ + far];
    float best = -1.f; int bi = 0;
    for (int n = t; n < N_; n += blockDim.x) {
      float dx = xb[n] - cx, dy = xb[N_ + n] - cy, dz = xb[2 * N_ + n] - cz;
      float d = dx * dx + dy * dy + dz * dz;
      float dm = fminf(dist[n], d);
      dist[n] = dm;
      if (dm > best) { best = dm; bi = n; }
    }
    rv[t] = best; ri[t] = bi;
    __syncthreads();
    for (int off = blockDim.x >> 1; off > 0; off >>= 1) {
      if (t < off) {
        if (rv[t + off] > rv[t]) { rv[t] = rv[t + off]; ri[t] = ri[t + off]; }
      }
      __syncthreads();
    }
    if (t == 0) s_far = ri[0];
    __syncthreads();
  }
}

__global__ void gather_new_xyz(const float* __restrict__ xyz, const int* __restrict__ fps_idx,
                               float* __restrict__ new_xyz) {
  int i = blockIdx.x * blockDim.x + threadIdx.x;
  if (i >= BS_) return;
  int b = i / S_;
  int n = fps_idx[i];
  const float* xb = xyz + (size_t)b * 3 * N_;
  new_xyz[i * 3 + 0] = xb[n];
  new_xyz[i * 3 + 1] = xb[N_ + n];
  new_xyz[i * 3 + 2] = xb[2 * N_ + n];
}

// ---------------- kNN (K smallest) : 1 block per center, dists in LDS -------
__global__ void knn_kernel(const float* __restrict__ xyz, const float* __restrict__ new_xyz,
                           int* __restrict__ knn_idx) {
  int bs = blockIdx.x;
  int b  = bs / S_;
  __shared__ float dist[N_];
  __shared__ float rv[256];
  __shared__ int   ri[256];
  const float* xb = xyz + (size_t)b * 3 * N_;
  float cx = new_xyz[bs * 3], cy = new_xyz[bs * 3 + 1], cz = new_xyz[bs * 3 + 2];
  int t = threadIdx.x;
  for (int n = t; n < N_; n += 256) {
    float dx = xb[n] - cx, dy = xb[N_ + n] - cy, dz = xb[2 * N_ + n] - cz;
    dist[n] = dx * dx + dy * dy + dz * dz;
  }
  __syncthreads();
  for (int kk = 0; kk < K_; ++kk) {
    float best = 3.4e38f; int bi = N_;
    for (int n = t; n < N_; n += 256) {
      float d = dist[n];
      if (d < best || (d == best && n < bi)) { best = d; bi = n; }
    }
    rv[t] = best; ri[t] = bi;
    __syncthreads();
    for (int off = 128; off > 0; off >>= 1) {
      if (t < off) {
        if (rv[t + off] < rv[t] || (rv[t + off] == rv[t] && ri[t + off] < ri[t])) {
          rv[t] = rv[t + off]; ri[t] = ri[t + off];
        }
      }
      __syncthreads();
    }
    if (t == 0) { knn_idx[bs * K_ + kk] = ri[0]; dist[ri[0]] = 3.4e38f; }
    __syncthreads();
  }
}

// ---------------- grouped feature build -> fp16 [M, 160] --------------------
__global__ void build_feats(const float* __restrict__ xyz, const float* __restrict__ points,
                            const float* __restrict__ new_xyz, const int* __restrict__ fps_idx,
                            const int* __restrict__ knn_idx, _Float16* __restrict__ feats) {
  int r  = blockIdx.x;            // [0, M)
  int bs = r / K_;
  int b  = bs / S_;
  int g  = knn_idx[r];
  int c0 = fps_idx[bs];
  const float* xb = xyz    + (size_t)b * 3 * N_;
  const float* pb = points + (size_t)b * D_ * N_;
  _Float16* out = feats + (size_t)r * CP1_;
  int t = threadIdx.x;
  if (t < 3) {
    out[t] = (_Float16)(xb[t * N_ + g] - new_xyz[bs * 3 + t]);
  } else if (t < 3 + D_) {
    int d = t - 3;
    out[t] = (_Float16)pb[(size_t)d * N_ + c0];
  } else if (t < 3 + 2 * D_) {
    int d = t - 3 - D_;
    out[t] = (_Float16)(pb[(size_t)d * N_ + g] - pb[(size_t)d * N_ + c0]);
  } else if (t < CP1_) {
    out[t] = (_Float16)0.f;
  }
}

// ---------------- WMMA fp16 GEMM: C[M,N] = A[M,K] * Wp + bias ---------------
// one wave per 16x64 strip (4 accumulators); A: two b128 loads/lane/k-step,
// B: fragment-packed, one v16h (2x b128) load per column tile.
__global__ void wmma_gemm(const _Float16* __restrict__ A, const _Float16* __restrict__ Bp,
                          const float* __restrict__ bias, float* __restrict__ C,
                          int Mr, int Nc, int Kc) {
  int wave = threadIdx.x >> 5;
  int lane = threadIdx.x & 31;
  int tilesN4 = Nc >> 6;
  int nTiles = (Mr >> 4) * tilesN4;
  int tile = blockIdx.x * 8 + wave;
  if (tile >= nTiles) return;                 // wave-uniform: EXEC stays all-ones
  int tm = tile / tilesN4, tn4 = tile % tilesN4;
  int row0 = tm << 4;
  int m = lane & 15, h = lane >> 4;
  int tilesN = Nc >> 4;
  const _Float16* arow = A + (size_t)(row0 + m) * Kc + h * 16;
  v8f acc0 = {}, acc1 = {}, acc2 = {}, acc3 = {};
  for (int k0 = 0; k0 < Kc; k0 += 32) {
    v8h lo = *(const v8h*)(arow + k0);
    v8h hi = *(const v8h*)(arow + k0 + 8);
    v16h a = __builtin_shufflevector(lo, hi, 0, 1, 2, 3, 4, 5, 6, 7,
                                             8, 9, 10, 11, 12, 13, 14, 15);
    __builtin_prefetch(arow + k0 + 32, 0, 1);   // next A k-block
    const _Float16* bbase = Bp + ((size_t)(k0 >> 5) * tilesN + (tn4 << 2)) * 512 + lane * 16;
    v16h b0 = *(const v16h*)(bbase);
    v16h b1 = *(const v16h*)(bbase + 512);
    v16h b2 = *(const v16h*)(bbase + 1024);
    v16h b3 = *(const v16h*)(bbase + 1536);
    acc0 = __builtin_amdgcn_wmma_f32_16x16x32_f16(false, a, false, b0, (short)0, acc0, false, false);
    acc1 = __builtin_amdgcn_wmma_f32_16x16x32_f16(false, a, false, b1, (short)0, acc1, false, false);
    acc2 = __builtin_amdgcn_wmma_f32_16x16x32_f16(false, a, false, b2, (short)0, acc2, false, false);
    acc3 = __builtin_amdgcn_wmma_f32_16x16x32_f16(false, a, false, b3, (short)0, acc3, false, false);
  }
  float bv0 = 0.f, bv1 = 0.f, bv2 = 0.f, bv3 = 0.f;
  if (bias) {
    const float* bp = bias + (tn4 << 6) + m;
    bv0 = bp[0]; bv1 = bp[16]; bv2 = bp[32]; bv3 = bp[48];
  }
  float* crow = C + (size_t)(row0 + h * 8) * Nc + (tn4 << 6) + m;
#pragma unroll
  for (int r = 0; r < 8; ++r) {
    crow[0]  = acc0[r] + bv0;
    crow[16] = acc1[r] + bv1;
    crow[32] = acc2[r] + bv2;
    crow[48] = acc3[r] + bv3;
    crow += Nc;
  }
}

// ---------------- BatchNorm (training stats over M rows) --------------------
__global__ void bn_stats(const float* __restrict__ X, float* __restrict__ stats,
                         int Mrows, int C) {
  int c = blockIdx.x;
  __shared__ float s1[256], s2[256];
  float a = 0.f, b2 = 0.f;
  for (int r = threadIdx.x; r < Mrows; r += 256) {
    float v = X[(size_t)r * C + c];
    a += v; b2 += v * v;
  }
  s1[threadIdx.x] = a; s2[threadIdx.x] = b2;
  __syncthreads();
  for (int off = 128; off > 0; off >>= 1) {
    if (threadIdx.x < off) {
      s1[threadIdx.x] += s1[threadIdx.x + off];
      s2[threadIdx.x] += s2[threadIdx.x + off];
    }
    __syncthreads();
  }
  if (threadIdx.x == 0) {
    float mu  = s1[0] / (float)Mrows;
    float var = s2[0] / (float)Mrows - mu * mu;
    stats[c]     = mu;
    stats[C + c] = rsqrtf(var + 1e-5f);
  }
}

__global__ void bn_norm_relu(const float* __restrict__ X, const float* __restrict__ stats,
                             const float* __restrict__ gamma, const float* __restrict__ beta,
                             _Float16* __restrict__ Y16, float* __restrict__ Y32,
                             int Mrows, int C) {
  size_t i = (size_t)blockIdx.x * blockDim.x + threadIdx.x;
  size_t total = (size_t)Mrows * C;
  if (i >= total) return;
  int c = (int)(i % C);
  float v = X[i];
  v = gamma[c] * (v - stats[c]) * stats[C + c] + beta[c];
  v = fmaxf(v, 0.f);
  if (Y16) Y16[i] = (_Float16)v;
  if (Y32) Y32[i] = v;
}

// ---------------- max over K neighbors --------------------------------------
__global__ void max_over_k(const float* __restrict__ H, float* __restrict__ feat,
                           _Float16* __restrict__ feat16) {
  int i = blockIdx.x * blockDim.x + threadIdx.x;
  if (i >= BS_ * DM_) return;
  int r = i / DM_, c = i % DM_;
  float m = -3.4e38f;
  for (int k = 0; k < K_; ++k)
    m = fmaxf(m, H[((size_t)r * K_ + k) * DM_ + c]);
  feat[i] = m;
  feat16[i] = (_Float16)m;
}

// ---------------- small elementwise helpers ---------------------------------
__global__ void build_pos_in(const float* __restrict__ new_xyz, _Float16* __restrict__ A) {
  int i = blockIdx.x * blockDim.x + threadIdx.x;
  if (i >= BS_ * 32) return;
  int r = i >> 5, c = i & 31;
  A[i] = (c < 3) ? (_Float16)new_xyz[r * 3 + c] : (_Float16)0.f;
}

__global__ void relu_f16(const float* __restrict__ X, _Float16* __restrict__ Y, int total) {
  int i = blockIdx.x * blockDim.x + threadIdx.x;
  if (i >= total) return;
  Y[i] = (_Float16)fmaxf(X[i], 0.f);
}

__global__ void add_pair(const float* __restrict__ A, const float* __restrict__ Bv,
                         float* __restrict__ Y32, _Float16* __restrict__ Y16, int total) {
  int i = blockIdx.x * blockDim.x + threadIdx.x;
  if (i >= total) return;
  float v = A[i] + Bv[i];
  Y32[i] = v;
  Y16[i] = (_Float16)v;
}

__global__ void elu1_scale(const float* __restrict__ Q, const float* __restrict__ Kk,
                           const float* __restrict__ V, float* __restrict__ Qe,
                           float* __restrict__ Ke, float* __restrict__ Vs,
                           float invS, int total) {
  int i = blockIdx.x * blockDim.x + threadIdx.x;
  if (i >= total) return;
  float q = Q[i];  Qe[i] = (q > 0.f) ? q + 1.f : expf(q);
  float k = Kk[i]; Ke[i] = (k > 0.f) ? k + 1.f : expf(k);
  Vs[i] = V[i] * invS;
}

__global__ void ksum_kernel(const float* __restrict__ Ke, float* __restrict__ Ksum) {
  int i = blockIdx.x * blockDim.x + threadIdx.x;
  if (i >= B_ * DM_) return;
  int b = i / DM_, c = i % DM_;
  float s = 0.f;
  for (int ss = 0; ss < S_; ++ss) s += Ke[((size_t)b * S_ + ss) * DM_ + c];
  Ksum[i] = s;
}

// KV[b,h,d,v] = sum_s Ke[b,s,h,d] * Vs[b,s,h,v]; block per (b,h,d), 64 threads (v)
__global__ void kv_kernel(const float* __restrict__ Ke, const float* __restrict__ Vs,
                          float* __restrict__ KV) {
  int blk = blockIdx.x;
  int d = blk % HD_;
  int h = (blk / HD_) % NH_;
  int b = blk / (HD_ * NH_);
  int v = threadIdx.x;
  float acc = 0.f;
  for (int s = 0; s < S_; ++s) {
    size_t base = ((size_t)b * S_ + s) * DM_ + h * HD_;
    acc += Ke[base + d] * Vs[base + v];
  }
  KV[(((size_t)b * NH_ + h) * HD_ + d) * HD_ + v] = acc;
}

// msg[b,s,h,v] = (Q . KV[:,v]) * S / (Q . Ksum + eps); block per (b,s,h)
__global__ void msg_kernel(const float* __restrict__ Qe, const float* __restrict__ KV,
                           const float* __restrict__ Ksum, float* __restrict__ Msg,
                           _Float16* __restrict__ Msg16) {
  int blk = blockIdx.x;
  int h  = blk % NH_;
  int bs = blk / NH_;
  int b  = bs / S_;
  int v  = threadIdx.x;
  const float* qrow = Qe + (size_t)bs * DM_ + h * HD_;
  const float* ks   = Ksum + ((size_t)b * NH_ + h) * HD_;
  const float* kv   = KV + ((size_t)b * NH_ + h) * HD_ * HD_;
  __shared__ float qs[HD_];
  qs[v] = qrow[v];
  __syncthreads();
  float zden = 0.f, acc = 0.f;
  for (int d = 0; d < HD_; ++d) {
    zden += qs[d] * ks[d];
    acc  += qs[d] * kv[d * HD_ + v];
  }
  float val = acc * (1.f / (zden + 1e-6f)) * (float)S_;
  size_t o = (size_t)bs * DM_ + h * HD_ + v;
  Msg[o] = val;
  Msg16[o] = (_Float16)val;
}

// ---------------- layernorm over last dim (C == 256), 1 row per block -------
__global__ void layernorm(const float* __restrict__ X, const float* __restrict__ g,
                          const float* __restrict__ bta, float* __restrict__ Y) {
  int r = blockIdx.x;
  int t = threadIdx.x;
  __shared__ float s1[256], s2[256];
  float v = X[(size_t)r * DM_ + t];
  s1[t] = v; s2[t] = v * v;
  __syncthreads();
  for (int off = 128; off > 0; off >>= 1) {
    if (t < off) { s1[t] += s1[t + off]; s2[t] += s2[t + off]; }
    __syncthreads();
  }
  float mu  = s1[0] / (float)DM_;
  float var = s2[0] / (float)DM_ - mu * mu;
  Y[(size_t)r * DM_ + t] = g[t] * (v - mu) * rsqrtf(var + 1e-5f) + bta[t];
}

__global__ void concat_kernel(const float* __restrict__ feat, const float* __restrict__ msg,
                              _Float16* __restrict__ cat) {
  int i = blockIdx.x * blockDim.x + threadIdx.x;
  if (i >= BS_ * 2 * DM_) return;
  int r = i / (2 * DM_), c = i % (2 * DM_);
  float v = (c < DM_) ? feat[(size_t)r * DM_ + c] : msg[(size_t)r * DM_ + (c - DM_)];
  cat[i] = (_Float16)v;
}

__global__ void final_out(const float* __restrict__ feat, const float* __restrict__ msg2,
                          float* __restrict__ out) {
  int i = blockIdx.x * blockDim.x + threadIdx.x;
  if (i >= B_ * DM_ * S_) return;
  int s = i % S_;
  int c = (i / S_) % DM_;
  int b = i / (S_ * DM_);
  size_t bs = (size_t)b * S_ + s;
  out[i] = feat[bs * DM_ + c] + msg2[bs * DM_ + c];
}

// ============================ host launcher =================================
extern "C" void kernel_launch(void* const* d_in, const int* in_sizes, int n_in,
                              void* d_out, int out_size, void* d_ws, size_t ws_size,
                              hipStream_t stream) {
  (void)in_sizes; (void)n_in; (void)out_size; (void)ws_size;
  const float* xyz    = (const float*)d_in[0];
  const float* points = (const float*)d_in[1];
  const float* W1  = (const float*)d_in[2];
  const float* b1  = (const float*)d_in[3];
  const float* g1  = (const float*)d_in[4];
  const float* be1 = (const float*)d_in[5];
  const float* W2  = (const float*)d_in[6];
  const float* b2  = (const float*)d_in[7];
  const float* g2  = (const float*)d_in[8];
  const float* be2 = (const float*)d_in[9];
  const float* W3  = (const float*)d_in[10];
  const float* b3  = (const float*)d_in[11];
  const float* g3  = (const float*)d_in[12];
  const float* be3 = (const float*)d_in[13];
  const float* pos_w1 = (const float*)d_in[14];
  const float* pos_b1 = (const float*)d_in[15];
  const float* pos_w2 = (const float*)d_in[16];
  const float* pos_b2 = (const float*)d_in[17];
  const float* q_w    = (const float*)d_in[18];
  const float* k_w    = (const float*)d_in[19];
  const float* v_w    = (const float*)d_in[20];
  const float* merge_w= (const float*)d_in[21];
  const float* mlp_w1 = (const float*)d_in[22];
  const float* mlp_w2 = (const float*)d_in[23];
  const float* ln1_g  = (const float*)d_in[24];
  const float* ln1_b  = (const float*)d_in[25];
  const float* ln2_g  = (const float*)d_in[26];
  const float* ln2_b  = (const float*)d_in[27];

  char* ws = (char*)d_ws;
  size_t off = 0;
  auto alloc = [&](size_t bytes) -> void* {
    void* p = ws + off;
    off += (bytes + 255) & ~(size_t)255;
    return p;
  };

  // --- big regions (reused across phases) ---
  _Float16* feats = (_Float16*)alloc((size_t)M_ * CP1_ * 2);   // 160 MB; attn fp16 temps later
  float*    Hbuf  = (float*)alloc((size_t)M_ * 256 * 4);       // 512 MB; attn fp32 temps later
  _Float16* h16   = (_Float16*)alloc((size_t)M_ * 128 * 2);    // 128 MB; feat fp32 later

  // --- small persistent buffers ---
  int*   fpsIdx = (int*)alloc((size_t)BS_ * 4);
  int*   knnIdx = (int*)alloc((size_t)BS_ * K_ * 4);
  float* newXyz = (float*)alloc((size_t)BS_ * 3 * 4);
  float* stats  = (float*)alloc(512 * 4);
  _Float16* w1h = (_Float16*)alloc(160 * 128 * 2);
  _Float16* w2h = (_Float16*)alloc(128 * 128 * 2);
  _Float16* w3h = (_Float16*)alloc(128 * 256 * 2);
  _Float16* pw1h= (_Float16*)alloc(32 * 256 * 2);
  _Float16* pw2h= (_Float16*)alloc(256 * 256 * 2);
  _Float16* qwh = (_Float16*)alloc(256 * 256 * 2);
  _Float16* kwh = (_Float16*)alloc(256 * 256 * 2);
  _Float16* vwh = (_Float16*)alloc(256 * 256 * 2);
  _Float16* mwh = (_Float16*)alloc(256 * 256 * 2);
  _Float16* m1h = (_Float16*)alloc(512 * 512 * 2);
  _Float16* m2h = (_Float16*)alloc(512 * 256 * 2);

  // --- attention-phase aliases ---
  const size_t SZ = (size_t)BS_ * DM_;           // 4 M floats
  float* feat  = (float*)h16;                    // h16 region is free after GEMM3
  float* pos1  = Hbuf + 0 * SZ;                  // Hbuf free after max_over_k
  float* pos2  = Hbuf + 1 * SZ;
  float* fpsum = Hbuf + 2 * SZ;
  float* qb    = Hbuf + 3 * SZ;
  float* kb    = Hbuf + 4 * SZ;
  float* vb    = Hbuf + 5 * SZ;
  float* Qe    = Hbuf + 6 * SZ;
  float* Ke    = Hbuf + 7 * SZ;
  float* Vs    = Hbuf + 8 * SZ;
  float* msg   = Hbuf + 9 * SZ;
  float* msgm  = Hbuf + 10 * SZ;
  float* ln1   = Hbuf + 11 * SZ;
  float* mlp1  = Hbuf + 12 * SZ;                 // BS x 512 -> 2*SZ
  float* msg2  = Hbuf + 14 * SZ;
  float* ln2   = Hbuf + 15 * SZ;
  float* KV    = Hbuf + 16 * SZ;                 // B*NH*64*64
  float* Ksum  = KV + (size_t)B_ * NH_ * HD_ * HD_;
  _Float16* feat16  = feats;                     // feats region free after GEMM1
  _Float16* posA16  = feat16 + SZ;               // BS x 32
  _Float16* pos1_16 = posA16 + (size_t)BS_ * 32;
  _Float16* fp16b   = pos1_16 + SZ;
  _Float16* msg16   = fp16b + SZ;
  _Float16* cat16   = msg16 + SZ;                // BS x 512
  _Float16* mlp1_16 = cat16 + 2 * SZ;            // BS x 512

  auto gemm = [&](const _Float16* A, const _Float16* W, const float* bias, float* C,
                  int Mr, int Nc, int Kc) {
    int nTiles = (Mr / 16) * (Nc / 64);
    wmma_gemm<<<dim3((nTiles + 7) / 8), dim3(256), 0, stream>>>(A, W, bias, C, Mr, Nc, Kc);
  };
  auto cvt = [&](const float* W, _Float16* Wh, int rows, int rowsPad, int cols) {
    int total = rowsPad * cols;
    cvt_w_pack<<<dim3((total + 255) / 256), dim3(256), 0, stream>>>(W, Wh, rows, rowsPad, cols);
  };

  // ---- weight repack into WMMA fragment order ----
  cvt(W1, w1h, 131, 160, 128);
  cvt(W2, w2h, 128, 128, 128);
  cvt(W3, w3h, 128, 128, 256);
  cvt(pos_w1, pw1h, 3, 32, 256);
  cvt(pos_w2, pw2h, 256, 256, 256);
  cvt(q_w, qwh, 256, 256, 256);
  cvt(k_w, kwh, 256, 256, 256);
  cvt(v_w, vwh, 256, 256, 256);
  cvt(merge_w, mwh, 256, 256, 256);
  cvt(mlp_w1, m1h, 512, 512, 512);
  cvt(mlp_w2, m2h, 512, 512, 256);

  // ---- sampling / grouping ----
  fps_kernel<<<dim3(B_), dim3(1024), 0, stream>>>(xyz, fpsIdx);
  gather_new_xyz<<<dim3((BS_ + 255) / 256), dim3(256), 0, stream>>>(xyz, fpsIdx, newXyz);
  knn_kernel<<<dim3(BS_), dim3(256), 0, stream>>>(xyz, newXyz, knnIdx);
  build_feats<<<dim3(M_), dim3(CP1_), 0, stream>>>(xyz, points, newXyz, fpsIdx, knnIdx, feats);

  // ---- conv MLP stack (WMMA GEMM + BN(train) + ReLU) ----
  gemm(feats, w1h, b1, Hbuf, M_, 128, 160);
  bn_stats<<<dim3(128), dim3(256), 0, stream>>>(Hbuf, stats, M_, 128);
  bn_norm_relu<<<dim3((int)(((size_t)M_ * 128 + 255) / 256)), dim3(256), 0, stream>>>(
      Hbuf, stats, g1, be1, h16, nullptr, M_, 128);
  gemm(h16, w2h, b2, Hbuf, M_, 128, 128);
  bn_stats<<<dim3(128), dim3(256), 0, stream>>>(Hbuf, stats, M_, 128);
  bn_norm_relu<<<dim3((int)(((size_t)M_ * 128 + 255) / 256)), dim3(256), 0, stream>>>(
      Hbuf, stats, g2, be2, h16, nullptr, M_, 128);
  gemm(h16, w3h, b3, Hbuf, M_, 256, 128);
  bn_stats<<<dim3(256), dim3(256), 0, stream>>>(Hbuf, stats, M_, 256);
  bn_norm_relu<<<dim3((int)(((size_t)M_ * 256 + 255) / 256)), dim3(256), 0, stream>>>(
      Hbuf, stats, g3, be3, nullptr, Hbuf, M_, 256);
  max_over_k<<<dim3((BS_ * DM_ + 255) / 256), dim3(256), 0, stream>>>(Hbuf, feat, feat16);

  // ---- position encoding MLP ----
  build_pos_in<<<dim3((BS_ * 32 + 255) / 256), dim3(256), 0, stream>>>(newXyz, posA16);
  gemm(posA16, pw1h, pos_b1, pos1, BS_, 256, 32);
  relu_f16<<<dim3((BS_ * 256 + 255) / 256), dim3(256), 0, stream>>>(pos1, pos1_16, BS_ * 256);
  gemm(pos1_16, pw2h, pos_b2, pos2, BS_, 256, 256);
  add_pair<<<dim3((BS_ * 256 + 255) / 256), dim3(256), 0, stream>>>(feat, pos2, fpsum, fp16b, BS_ * 256);

  // ---- linear self-attention ----
  gemm(fp16b, qwh, nullptr, qb, BS_, 256, 256);
  gemm(fp16b, kwh, nullptr, kb, BS_, 256, 256);
  gemm(fp16b, vwh, nullptr, vb, BS_, 256, 256);
  elu1_scale<<<dim3((BS_ * 256 + 255) / 256), dim3(256), 0, stream>>>(
      qb, kb, vb, Qe, Ke, Vs, 1.f / (float)S_, BS_ * 256);
  ksum_kernel<<<dim3((B_ * DM_ + 255) / 256), dim3(256), 0, stream>>>(Ke, Ksum);
  kv_kernel<<<dim3(B_ * NH_ * HD_), dim3(64), 0, stream>>>(Ke, Vs, KV);
  msg_kernel<<<dim3(BS_ * NH_), dim3(64), 0, stream>>>(Qe, KV, Ksum, msg, msg16);
  gemm(msg16, mwh, nullptr, msgm, BS_, 256, 256);
  layernorm<<<dim3(BS_), dim3(256), 0, stream>>>(msgm, ln1_g, ln1_b, ln1);

  // ---- post MLP + residual ----
  concat_kernel<<<dim3((BS_ * 512 + 255) / 256), dim3(256), 0, stream>>>(feat, ln1, cat16);
  gemm(cat16, m1h, nullptr, mlp1, BS_, 512, 512);
  relu_f16<<<dim3((BS_ * 512 + 255) / 256), dim3(256), 0, stream>>>(mlp1, mlp1_16, BS_ * 512);
  gemm(mlp1_16, m2h, nullptr, msg2, BS_, 256, 512);
  layernorm<<<dim3(BS_), dim3(256), 0, stream>>>(msg2, ln2_g, ln2_b, ln2);
  final_out<<<dim3((B_ * DM_ * S_ + 255) / 256), dim3(256), 0, stream>>>(feat, ln2, (float*)d_out);
}
